// HandNet_2302102471380
// MI455X (gfx1250) — compile-verified, matrix operand order checked by reading, not run on hardware
//
#include <hip/hip_runtime.h>

typedef __attribute__((ext_vector_type(16))) __bf16 v16bf;
typedef __attribute__((ext_vector_type(8)))  float  v8f;
typedef __attribute__((ext_vector_type(4)))  int    v4i;

#define AS1 __attribute__((address_space(1)))
#define AS3 __attribute__((address_space(3)))

#define NUM_HAND 21
#define BATCH    8192
#define M_ROWS   (BATCH * NUM_HAND)   // 172032, divisible by 128

// ---------------------------------------------------------------------------
// CDNA5 async global->LDS copy (ASYNCcnt path), signature probe-confirmed:
//   __builtin_amdgcn_global_load_async_to_lds_b128(v4i AS1*, v4i AS3*, imm, imm)
// ---------------------------------------------------------------------------
#if __has_builtin(__builtin_amdgcn_global_load_async_to_lds_b128)
#define HAS_ASYNC_LDS 1
#else
#define HAS_ASYNC_LDS 0
#endif

template <int OFF>
__device__ __forceinline__ void async_copy_b128(const __bf16* gsrc, __bf16* ldst) {
#if HAS_ASYNC_LDS
  __builtin_amdgcn_global_load_async_to_lds_b128(
      (AS1 v4i*)gsrc, (AS3 v4i*)ldst, OFF, 0);
#else
  *(uint4*)((char*)ldst + OFF) = *(const uint4*)((const char*)gsrc + OFF);
#endif
}

template <int N>
__device__ __forceinline__ void wait_async() {
#if HAS_ASYNC_LDS
#if __has_builtin(__builtin_amdgcn_s_wait_asynccnt)
  __builtin_amdgcn_s_wait_asynccnt(N);
#else
  asm volatile("s_wait_asynccnt %0" :: "i"(N) : "memory");
#endif
#endif
}

// ---------------------------------------------------------------------------
// Weight prep: convert f32 [K][N] row-major -> bf16 [N][K] (transposed) so a
// WMMA B-fragment (column N, 16 consecutive K) is a contiguous 32B LDS read.
// ---------------------------------------------------------------------------
__global__ void prep_weights(const float* __restrict__ W1,
                             const float* __restrict__ W2,
                             const float* __restrict__ W3,
                             __bf16* __restrict__ Wt1,
                             __bf16* __restrict__ Wt2,
                             __bf16* __restrict__ Wt3) {
  int idx = blockIdx.x * blockDim.x + threadIdx.x;
  if (idx < 256 * 128) {                 // W1 [256][128] -> Wt1 [128][256]
    int k = idx >> 7, n = idx & 127;
    Wt1[n * 256 + k] = (__bf16)W1[idx];
  }
  if (idx < 128 * 128) {                 // W2 [128][128] -> Wt2 [128][128]
    int k = idx >> 7, n = idx & 127;
    Wt2[n * 128 + k] = (__bf16)W2[idx];
  }
  if (idx < 128 * 512) {                 // W3 [128][512] -> Wt3 [512][128]
    int k = idx >> 9, n = idx & 511;
    Wt3[n * 128 + k] = (__bf16)W3[idx];
  }
}

// ---------------------------------------------------------------------------
// Mix: Z[b,n,:] = sum_m A[n,m] * in[b,m,:]   (dense 21x21, staged in LDS)
// One batch item per workgroup; output bf16 for the WMMA GEMM.
// ---------------------------------------------------------------------------
template <typename TIN, int F>
__global__ void mix_kernel(const TIN* __restrict__ in,
                           const float* __restrict__ A,
                           __bf16* __restrict__ out) {
  __shared__ float sIn[NUM_HAND * F];
  __shared__ float sA[NUM_HAND * NUM_HAND];
  const int b = blockIdx.x;
  const TIN* inB = in + (size_t)b * NUM_HAND * F;
  for (int i = threadIdx.x; i < NUM_HAND * F; i += blockDim.x)
    sIn[i] = (float)inB[i];
  for (int i = threadIdx.x; i < NUM_HAND * NUM_HAND; i += blockDim.x)
    sA[i] = A[i];
  __syncthreads();
  __bf16* outB = out + (size_t)b * NUM_HAND * F;
  for (int i = threadIdx.x; i < NUM_HAND * F; i += blockDim.x) {
    const int n = i / F;
    const int f = i - n * F;
    float acc = 0.f;
#pragma unroll
    for (int m = 0; m < NUM_HAND; ++m)
      acc = fmaf(sA[n * NUM_HAND + m], sIn[m * F + f], acc);
    outB[i] = (__bf16)acc;
  }
}

// ---------------------------------------------------------------------------
// GEMM: Out = relu(Z @ Wt^T + bias)   via v_wmma_f32_16x16x32_bf16
//   Z  : [M_ROWS][K]  bf16 row-major
//   Wt : [Nfull][K]   bf16 (pre-transposed weights)
// Block: 256 threads (8 waves). Tile: 128(M) x 128(N), KT = 32.
// Wave w owns M-tile w and all 8 N-tiles (8 v8f accumulators = 64 VGPRs).
// Double-buffered LDS staging via async global->LDS copies (ASYNCcnt),
// pipelined one K-stage ahead of the WMMA compute.
// Each thread issues 4 async b128 ops per stage (2 for A, 2 for B), so
// s_wait_asynccnt 4 after issuing stage k+1 retires stage k.
// ---------------------------------------------------------------------------
template <int K, bool OUT_F32>
__global__ void gconv_gemm(const __bf16* __restrict__ Z,
                           const __bf16* __restrict__ Wt,
                           const float* __restrict__ bias,
                           void* __restrict__ outp, int Nfull) {
  constexpr int MB = 128, NB = 128, KT = 32, NK = K / KT;
  __shared__ alignas(32) __bf16 As[2][MB][KT];   // 2 x 8 KB
  __shared__ alignas(32) __bf16 Bs[2][NB][KT];   // 2 x 8 KB

  const int m0   = blockIdx.x * MB;
  const int n0   = blockIdx.y * NB;
  const int t    = threadIdx.x;
  const int wave = t >> 5;
  const int lane = t & 31;
  const int lrow = lane & 15;
  const int koff = (lane >> 4) << 4;          // 0 or 16

  // Per-thread staging assignment: 128 rows x 64B per tile, 32B per thread.
  const int srow = t >> 1, shalf = t & 1;
  const __bf16* gA = Z  + (size_t)(m0 + srow) * K + shalf * 16;
  const __bf16* gB = Wt + (size_t)(n0 + srow) * K + shalf * 16;

  auto issue_stage = [&](int kt, int buf) {
    const __bf16* a = gA + kt * KT;
    const __bf16* b = gB + kt * KT;
    __bf16* la = &As[buf][srow][shalf * 16];
    __bf16* lb = &Bs[buf][srow][shalf * 16];
    async_copy_b128<0>(a, la);
    async_copy_b128<16>(a, la);                    // +16B on both sides
    async_copy_b128<0>(b, lb);
    async_copy_b128<16>(b, lb);
  };

  v8f acc[8];
#pragma unroll
  for (int j = 0; j < 8; ++j)
    acc[j] = (v8f){0.f, 0.f, 0.f, 0.f, 0.f, 0.f, 0.f, 0.f};

  issue_stage(0, 0);
#pragma unroll
  for (int kt = 0; kt < NK; ++kt) {
    const int cur = kt & 1;
    if (kt + 1 < NK) {
      issue_stage(kt + 1, cur ^ 1);   // prefetch next stage into other buffer
      wait_async<4>();                // previous stage's 4 ops retired
    } else {
      wait_async<0>();
    }
    __syncthreads();                  // stage kt visible to all waves

    const v16bf a = *(const v16bf*)&As[cur][wave * 16 + lrow][koff];
#pragma unroll
    for (int j = 0; j < 8; ++j) {
      const v16bf bfr = *(const v16bf*)&Bs[cur][j * 16 + lrow][koff];
      acc[j] = __builtin_amdgcn_wmma_f32_16x16x32_bf16(
          /*neg_a=*/false, a, /*neg_b=*/false, bfr,
          /*c_mod=*/(short)0, acc[j], /*reuse_a=*/false, /*reuse_b=*/false);
    }
    __syncthreads();                  // all reads done before buffer reuse
  }

  // Epilogue: bias + ReLU, write bf16 (intermediate) or f32 (final x3).
  const int mbase = m0 + wave * 16 + ((lane >> 4) << 3);   // +8 for lanes>=16
#pragma unroll
  for (int j = 0; j < 8; ++j) {
    const int col = n0 + j * 16 + lrow;
    const float bv = bias[col];
#pragma unroll
    for (int r = 0; r < 8; ++r) {
      float v = acc[j][r] + bv;
      v = v > 0.f ? v : 0.f;
      const size_t idx = (size_t)(mbase + r) * Nfull + col;
      if constexpr (OUT_F32) ((float*)outp)[idx] = v;
      else                   ((__bf16*)outp)[idx] = (__bf16)v;
    }
  }
}

// ---------------------------------------------------------------------------
// FC: out[row,:] = x3[row,:] @ fcW + fcb  (512 -> 3). One wave per row,
// coalesced reads (lane-contiguous 128B chunks), shuffle reduction.
// ---------------------------------------------------------------------------
__global__ void fc_kernel(const float* __restrict__ x3,
                          const float* __restrict__ fcW,
                          const float* __restrict__ fcb,
                          float* __restrict__ out) {
  __shared__ float sW[512 * 3];
  for (int i = threadIdx.x; i < 512 * 3; i += blockDim.x) sW[i] = fcW[i];
  __syncthreads();
  const int wave = threadIdx.x >> 5;
  const int lane = threadIdx.x & 31;
  const int row  = blockIdx.x * 8 + wave;     // M_ROWS = 21504*8 exactly
  const float* xr = x3 + (size_t)row * 512;
  float a0 = 0.f, a1 = 0.f, a2 = 0.f;
#pragma unroll
  for (int i = 0; i < 16; ++i) {
    const int k = i * 32 + lane;
    const float xv = xr[k];
    a0 = fmaf(xv, sW[k * 3 + 0], a0);
    a1 = fmaf(xv, sW[k * 3 + 1], a1);
    a2 = fmaf(xv, sW[k * 3 + 2], a2);
  }
#pragma unroll
  for (int off = 16; off > 0; off >>= 1) {
    a0 += __shfl_down(a0, off);
    a1 += __shfl_down(a1, off);
    a2 += __shfl_down(a2, off);
  }
  if (lane == 0) {
    out[(size_t)row * 3 + 0] = a0 + fcb[0];
    out[(size_t)row * 3 + 1] = a1 + fcb[1];
    out[(size_t)row * 3 + 2] = a2 + fcb[2];
  }
}

// ---------------------------------------------------------------------------
extern "C" void kernel_launch(void* const* d_in, const int* in_sizes, int n_in,
                              void* d_out, int out_size, void* d_ws, size_t ws_size,
                              hipStream_t stream) {
  (void)in_sizes; (void)n_in; (void)out_size; (void)ws_size;
  // Inputs (setup_inputs order)
  const float* x   = (const float*)d_in[0];
  const float* A1  = (const float*)d_in[1];
  const float* A2  = (const float*)d_in[2];
  const float* A3  = (const float*)d_in[3];
  const float* W1  = (const float*)d_in[4];
  const float* b1  = (const float*)d_in[5];
  const float* W2  = (const float*)d_in[6];
  const float* b2  = (const float*)d_in[7];
  const float* W3  = (const float*)d_in[8];
  const float* b3  = (const float*)d_in[9];
  const float* fcW = (const float*)d_in[10];
  const float* fcb = (const float*)d_in[11];

  // Outputs: x3 [8192*21*512] f32, then out [8192*21*3] f32
  float* x3  = (float*)d_out;
  float* fco = (float*)d_out + (size_t)M_ROWS * 512;

  // Workspace layout
  char* ws = (char*)d_ws;
  __bf16* Wt1  = (__bf16*)(ws);                               // 64 KB
  __bf16* Wt2  = (__bf16*)(ws + 65536);                       // 32 KB
  __bf16* Wt3  = (__bf16*)(ws + 98304);                       // 128 KB
  __bf16* Zbuf = (__bf16*)(ws + 262144);                      // 88,080,384 B
  __bf16* Ybuf = (__bf16*)(ws + 262144 + 88080384ull);        // 44,040,192 B

  prep_weights<<<256, 256, 0, stream>>>(W1, W2, W3, Wt1, Wt2, Wt3);

  // Layer 1: Z = A1 @ x ; Y1 = relu(Z @ W1 + b1)
  mix_kernel<float, 256><<<BATCH, 256, 0, stream>>>(x, A1, Zbuf);
  gconv_gemm<256, false><<<dim3(M_ROWS / 128, 128 / 128), 256, 0, stream>>>(
      Zbuf, Wt1, b1, (void*)Ybuf, 128);

  // Layer 2
  mix_kernel<__bf16, 128><<<BATCH, 256, 0, stream>>>(Ybuf, A2, Zbuf);
  gconv_gemm<128, false><<<dim3(M_ROWS / 128, 128 / 128), 256, 0, stream>>>(
      Zbuf, Wt2, b2, (void*)Ybuf, 128);

  // Layer 3 (writes x3 f32 directly to d_out)
  mix_kernel<__bf16, 128><<<BATCH, 256, 0, stream>>>(Ybuf, A3, Zbuf);
  gconv_gemm<128, true><<<dim3(M_ROWS / 128, 512 / 128), 256, 0, stream>>>(
      Zbuf, Wt3, b3, (void*)x3, 512);

  // FC head
  fc_kernel<<<M_ROWS / 8, 256, 0, stream>>>(x3, fcW, fcb, fco);
}
